// SpaceTempGoG_detr_dad_14250701488149
// MI455X (gfx1250) — matrix operand order repaired and checked by reading, not compiled, not run on hardware
//
#include <hip/hip_runtime.h>
#include <math.h>

// ---------------------------------------------------------------------------
// SpaceTempGoG forward for MI455X (gfx1250).
// - K=2048 input projections: operands pre-converted to bf16 in HBM once,
//   then staged with GLOBAL_LOAD_ASYNC_TO_LDS_B128 (ASYNCcnt) into LDS and
//   consumed by v_wmma_f32_16x16x32_bf16.
// - All other GEMMs: fp32 operands converted to bf16 in-register while
//   staging (128-bit loads/stores, register double buffering).
// - Edge ops: fp32 gather/scatter with global atomics (HBM/L2-bound).
// ---------------------------------------------------------------------------

#define NHEADS   4
#define N_FRAMES 256
#define NPF      50
#define POOL_KP  40
#define N_NODES  (N_FRAMES * NPF)
#define E_SP     204800
#define E_TM     204800
#define E_VID    1024
#define INPUT_D  2048
#define OBJ_D    300

typedef __attribute__((ext_vector_type(16))) __bf16 v16bf;
typedef __attribute__((ext_vector_type(8)))  __bf16 v8bf;
typedef __attribute__((ext_vector_type(8)))  float  v8f;
typedef __attribute__((ext_vector_type(4)))  float  v4f;
typedef __attribute__((ext_vector_type(4)))  unsigned v4u;

union BF16Frag { v16bf v; v8bf h[2]; };

__device__ __forceinline__ unsigned short f2bfu(float f) {
  unsigned u = __float_as_uint(f);
  u += 0x7FFFu + ((u >> 16) & 1u);          // round-to-nearest-even
  return (unsigned short)(u >> 16);
}
__device__ __forceinline__ unsigned pack2bf(float a, float b) {
  return (unsigned)f2bfu(a) | ((unsigned)f2bfu(b) << 16);
}

#define TBM 64
#define TBN 64
#define TBK 32
#define ASTR 40   // padded LDS row stride (bf16 elems); 80B keeps 16B alignment

// ---------------------------------------------------------------------------
// fp32 -> packed bf16 conversion (one pass; halves GEMM A re-read traffic)
// ---------------------------------------------------------------------------
__global__ void k_cvt_bf16(const float* __restrict__ X, int ldx,
                           unsigned short* __restrict__ Y, int ldy,
                           int M, int K) {
  size_t i = (size_t)blockIdx.x * blockDim.x + threadIdx.x;
  size_t tot = (size_t)M * (K >> 3);
  if (i >= tot) return;
  int r = (int)(i / (size_t)(K >> 3));
  int c = (int)(i % (size_t)(K >> 3)) * 8;
  const float* xp = X + (size_t)r * ldx + c;
  v4f f0 = *(const v4f*)xp;
  v4f f1 = *(const v4f*)(xp + 4);
  v4u p = { pack2bf(f0.x, f0.y), pack2bf(f0.z, f0.w),
            pack2bf(f1.x, f1.y), pack2bf(f1.z, f1.w) };
  *(v4u*)&Y[(size_t)r * ldy + c] = p;
}

// ---------------------------------------------------------------------------
// WMMA GEMM on pre-converted bf16 operands, async-DMA staging.
// C[M,N] = act( A_bf16[M,K] x B_bf16[N,K]^T + bias ).
// Requires M%64==0, N%64==0, K%32==0 (holds for all call sites).
// Tiles staged global->LDS via GLOBAL_LOAD_ASYNC_TO_LDS_B128; per-wave
// completion via s_wait_asynccnt, block-wide via s_barrier. Double buffered.
// ---------------------------------------------------------------------------
__global__ __launch_bounds__(256)
void k_gemm_bf16pre(const unsigned short* __restrict__ A, int lda,
                    const unsigned short* __restrict__ B, int ldb,
                    const float* __restrict__ bias,
                    float* __restrict__ C, int ldc,
                    int M, int N, int K, int act)
{
  __shared__ __align__(16) unsigned short As[2][TBM * ASTR];
  __shared__ __align__(16) unsigned short Bs[2][TBN * ASTR];
  const int tid  = threadIdx.x;
  const int lane = tid & 31;
  const int wave = tid >> 5;
  const int mB = blockIdx.x * TBM;
  const int nB = blockIdx.y * TBN;
  const int mw = (wave & 1) * 32;
  const int nw = (wave >> 1) * 16;
  const int l16 = lane & 15;
  const int hi  = lane >> 4;
  const int row = tid >> 2;            // 0..63
  const int ck8 = (tid & 3) * 8;       // 0,8,16,24 (bf16 elems; 16B chunks)

  auto issue = [&](int k0, int buf) {  // async DMA: 16B per lane for A and B
    const unsigned short* ga = A + (size_t)(mB + row) * lda + k0 + ck8;
    const unsigned short* gb = B + (size_t)(nB + row) * ldb + k0 + ck8;
    unsigned la = (unsigned)(size_t)&As[buf][row * ASTR + ck8];
    unsigned lb = (unsigned)(size_t)&Bs[buf][row * ASTR + ck8];
    asm volatile("global_load_async_to_lds_b128 %0, %1, off"
                 :: "v"(la), "v"(ga) : "memory");
    asm volatile("global_load_async_to_lds_b128 %0, %1, off"
                 :: "v"(lb), "v"(gb) : "memory");
  };

  v8f acc0 = {0.f, 0.f, 0.f, 0.f, 0.f, 0.f, 0.f, 0.f};
  v8f acc1 = acc0;

  auto compute = [&](int buf) {
    BF16Frag bf;
    const unsigned short* bsp = &Bs[buf][(nw + l16) * ASTR + hi * 8];
    bf.h[0] = *(const v8bf*)(bsp);
    bf.h[1] = *(const v8bf*)(bsp + 16);
    BF16Frag a0, a1;
    int kq = hi * 8;
    const unsigned short* p0 = &As[buf][(mw + l16) * ASTR];
    a0.h[0] = *(const v8bf*)(p0 + kq);
    a0.h[1] = *(const v8bf*)(p0 + kq + 16);
    const unsigned short* p1 = &As[buf][(mw + 16 + l16) * ASTR];
    a1.h[0] = *(const v8bf*)(p1 + kq);
    a1.h[1] = *(const v8bf*)(p1 + kq + 16);
    acc0 = __builtin_amdgcn_wmma_f32_16x16x32_bf16(false, a0.v, false, bf.v,
                                                   (short)0, acc0, false, false);
    acc1 = __builtin_amdgcn_wmma_f32_16x16x32_bf16(false, a1.v, false, bf.v,
                                                   (short)0, acc1, false, false);
  };

  issue(0, 0);
  int buf = 0;
  for (int k0 = 0; k0 < K; k0 += TBK) {
    // wait for this wave's copies into `buf` (oldest outstanding), then
    // barrier so every wave's chunk of `buf` has landed.
    asm volatile("s_wait_asynccnt 0" ::: "memory");
    __syncthreads();
    // safe: the barrier proves all waves finished reading buf^1 last iter.
    if (k0 + TBK < K) issue(k0 + TBK, buf ^ 1);   // DMA overlaps WMMAs below
    compute(buf);
    buf ^= 1;
  }

  const int col = nB + nw + l16;
  const float bv = bias ? bias[col] : 0.f;
  #pragma unroll
  for (int s = 0; s < 2; ++s) {
    v8f acc = s ? acc1 : acc0;
    int rbase = mB + mw + s * 16 + hi * 8;
    #pragma unroll
    for (int i = 0; i < 8; ++i) {
      float vv = acc[i] + bv;
      if (act == 1) vv = vv > 0.f ? vv : 0.2f * vv;
      else if (act == 2) vv = vv > 0.f ? vv : 0.f;
      C[(size_t)(rbase + i) * ldc + col] = vv;
    }
  }
}

// ---------------------------------------------------------------------------
// WMMA GEMM on fp32 operands (converted to bf16 while staging):
// C[M,N] = act( scale * A[M,K] x B + bias + add )
//   bNK==1 : B is [N,K] row-major (weights)   bNK==0 : B is [K,N] row-major
// Register-staged double buffering, 1 barrier / K-step.
// ---------------------------------------------------------------------------
__global__ __launch_bounds__(256)
void k_gemm_bf16(const float* __restrict__ A, int lda,
                 const float* __restrict__ B, int ldb, int bNK,
                 const float* __restrict__ bias,
                 const float* __restrict__ add, int ldadd,
                 float* __restrict__ C, int ldc, int colOff,
                 int M, int N, int K, float scale, int act)
{
  __shared__ __align__(16) unsigned short As[2][TBM * ASTR];
  __shared__ __align__(16) unsigned short Bs[2][TBN * ASTR];
  const int tid  = threadIdx.x;
  const int lane = tid & 31;
  const int wave = tid >> 5;
  const int mB = blockIdx.x * TBM;
  const int nB = blockIdx.y * TBN;
  const int mw = (wave & 1) * 32;
  const int nw = (wave >> 1) * 16;
  const int l16 = lane & 15;
  const int hi  = lane >> 4;

  const int amRow = tid >> 2;            // 0..63
  const int amCol = (tid & 3) * 8;       // 0,8,16,24
  const int btRow = tid >> 3;            // 0..31  (B[K,N] mode: k)
  const int btCol = (tid & 7) * 8;       // 0..56  (B[K,N] mode: n base)
  const bool aRows = (mB + TBM <= M);    // block-uniform interior tests
  const bool bCols = (nB + TBN <= N);

  float ra[8], rb[8];

  auto loadRegs = [&](int k0) {
    const bool kfull = (k0 + TBK <= K);
    if (aRows && kfull) {
      const float* ap = A + (size_t)(mB + amRow) * lda + k0 + amCol;
      v4f f0 = *(const v4f*)ap;
      v4f f1 = *(const v4f*)(ap + 4);
      ra[0] = f0.x; ra[1] = f0.y; ra[2] = f0.z; ra[3] = f0.w;
      ra[4] = f1.x; ra[5] = f1.y; ra[6] = f1.z; ra[7] = f1.w;
    } else {
      int gm = mB + amRow;
      #pragma unroll
      for (int i = 0; i < 8; ++i) {
        int gk = k0 + amCol + i;
        ra[i] = (gm < M && gk < K) ? A[(size_t)gm * lda + gk] : 0.f;
      }
    }
    if (bNK) {
      if (bCols && kfull) {
        const float* bp = B + (size_t)(nB + amRow) * ldb + k0 + amCol;
        v4f f0 = *(const v4f*)bp;
        v4f f1 = *(const v4f*)(bp + 4);
        rb[0] = f0.x; rb[1] = f0.y; rb[2] = f0.z; rb[3] = f0.w;
        rb[4] = f1.x; rb[5] = f1.y; rb[6] = f1.z; rb[7] = f1.w;
      } else {
        int gn = nB + amRow;
        #pragma unroll
        for (int i = 0; i < 8; ++i) {
          int gk = k0 + amCol + i;
          rb[i] = (gn < N && gk < K) ? B[(size_t)gn * ldb + gk] : 0.f;
        }
      }
    } else {
      if (bCols && kfull) {
        const float* bp = B + (size_t)(k0 + btRow) * ldb + nB + btCol;
        v4f f0 = *(const v4f*)bp;
        v4f f1 = *(const v4f*)(bp + 4);
        rb[0] = f0.x; rb[1] = f0.y; rb[2] = f0.z; rb[3] = f0.w;
        rb[4] = f1.x; rb[5] = f1.y; rb[6] = f1.z; rb[7] = f1.w;
      } else {
        int gk = k0 + btRow;
        #pragma unroll
        for (int i = 0; i < 8; ++i) {
          int gn = nB + btCol + i;
          rb[i] = (gn < N && gk < K) ? B[(size_t)gk * ldb + gn] : 0.f;
        }
      }
    }
  };

  auto storeRegs = [&](int buf) {
    v4u pa = { pack2bf(ra[0], ra[1]), pack2bf(ra[2], ra[3]),
               pack2bf(ra[4], ra[5]), pack2bf(ra[6], ra[7]) };
    *(v4u*)&As[buf][amRow * ASTR + amCol] = pa;           // ds_store_b128
    if (bNK) {
      v4u pb = { pack2bf(rb[0], rb[1]), pack2bf(rb[2], rb[3]),
                 pack2bf(rb[4], rb[5]), pack2bf(rb[6], rb[7]) };
      *(v4u*)&Bs[buf][amRow * ASTR + amCol] = pb;         // ds_store_b128
    } else {
      #pragma unroll
      for (int i = 0; i < 8; ++i)                         // LDS transpose scatter
        Bs[buf][(btCol + i) * ASTR + btRow] = f2bfu(rb[i]);
    }
  };

  v8f acc0 = {0.f, 0.f, 0.f, 0.f, 0.f, 0.f, 0.f, 0.f};
  v8f acc1 = acc0;

  auto compute = [&](int buf) {
    BF16Frag bf;
    const unsigned short* bsp = &Bs[buf][(nw + l16) * ASTR + hi * 8];
    bf.h[0] = *(const v8bf*)(bsp);
    bf.h[1] = *(const v8bf*)(bsp + 16);
    BF16Frag a0, a1;
    int kq = hi * 8;
    const unsigned short* p0 = &As[buf][(mw + l16) * ASTR];
    a0.h[0] = *(const v8bf*)(p0 + kq);
    a0.h[1] = *(const v8bf*)(p0 + kq + 16);
    const unsigned short* p1 = &As[buf][(mw + 16 + l16) * ASTR];
    a1.h[0] = *(const v8bf*)(p1 + kq);
    a1.h[1] = *(const v8bf*)(p1 + kq + 16);
    acc0 = __builtin_amdgcn_wmma_f32_16x16x32_bf16(false, a0.v, false, bf.v,
                                                   (short)0, acc0, false, false);
    acc1 = __builtin_amdgcn_wmma_f32_16x16x32_bf16(false, a1.v, false, bf.v,
                                                   (short)0, acc1, false, false);
  };

  loadRegs(0);
  storeRegs(0);
  int buf = 0;
  for (int k0 = 0; k0 < K; k0 += TBK) {
    const bool more = (k0 + TBK) < K;
    if (more) loadRegs(k0 + TBK);       // global loads overlap with WMMAs below
    __syncthreads();                    // stage `buf` visible to all waves
    compute(buf);
    if (more) storeRegs(buf ^ 1);
    buf ^= 1;
  }

  const int col = nB + nw + l16;
  const float bv = (bias != nullptr && col < N) ? bias[col] : 0.f;
  #pragma unroll
  for (int s = 0; s < 2; ++s) {
    v8f acc = s ? acc1 : acc0;
    int rbase = mB + mw + s * 16 + hi * 8;   // C layout: VGPR i -> M=i / M=8+i
    #pragma unroll
    for (int i = 0; i < 8; ++i) {
      int row = rbase + i;
      if (row < M && col < N) {
        float vv = acc[i] * scale + bv;
        if (add) vv += add[(size_t)row * ldadd + col];
        if (act == 1) vv = vv > 0.f ? vv : 0.2f * vv;        // LeakyReLU(0.2)
        else if (act == 2) vv = vv > 0.f ? vv : 0.f;          // ReLU
        C[(size_t)row * ldc + colOff + col] = vv;
      }
    }
  }
}

// ------------------------- elementwise / reduction -------------------------

__global__ void k_fill(float* __restrict__ p, size_t n, float v) {
  size_t i = (size_t)blockIdx.x * blockDim.x + threadIdx.x;
  if (i < n) p[i] = v;
}

// per-channel mean/var over M rows (biased var)
__global__ void k_col_stats(const float* __restrict__ X, int ldx, int M,
                            float* __restrict__ mean, float* __restrict__ var) {
  int c = blockIdx.x;
  __shared__ float ss[256], sq[256];
  float s = 0.f, q = 0.f;
  for (int r = threadIdx.x; r < M; r += blockDim.x) {
    float v = X[(size_t)r * ldx + c];
    s += v; q += v * v;
  }
  ss[threadIdx.x] = s; sq[threadIdx.x] = q;
  __syncthreads();
  for (int st = 128; st > 0; st >>= 1) {
    if (threadIdx.x < st) { ss[threadIdx.x] += ss[threadIdx.x + st]; sq[threadIdx.x] += sq[threadIdx.x + st]; }
    __syncthreads();
  }
  if (threadIdx.x == 0) {
    float m = ss[0] / (float)M;
    mean[c] = m;
    var[c] = sq[0] / (float)M - m * m;
  }
}

// y = act( (x-mean)*rsqrt(var+eps) [*g+b] ), write with ld/colOff
__global__ void k_norm_act(const float* __restrict__ X, int ldx,
                           const float* __restrict__ mean, const float* __restrict__ var,
                           const float* __restrict__ g, const float* __restrict__ b,
                           float* __restrict__ Y, int ldy, int colOff,
                           int M, int C, int act) {
  size_t i = (size_t)blockIdx.x * blockDim.x + threadIdx.x;
  if (i >= (size_t)M * C) return;
  int r = (int)(i / C), c = (int)(i % C);
  float v = (X[(size_t)r * ldx + c] - mean[c]) * rsqrtf(var[c] + 1e-5f);
  if (g) v = v * g[c] + b[c];
  if (act == 1) v = v > 0.f ? v : 0.2f * v;
  Y[(size_t)r * ldy + colOff + c] = v;
}

// ---------------------------- edge attention --------------------------------

__device__ __forceinline__ void atomicMaxF(float* addr, float val) {
  unsigned int* ia = (unsigned int*)addr;
  unsigned int old = *ia;
  while (__uint_as_float(old) < val) {
    unsigned int assumed = old;
    old = atomicCAS(ia, assumed, __float_as_uint(val));
    if (old == assumed) break;
  }
}

__global__ void k_edge_alpha(const float* __restrict__ qm, const float* __restrict__ km,
                             const int* __restrict__ src, const int* __restrict__ dst,
                             const float* __restrict__ escale, int estride,
                             const float* __restrict__ ew,
                             float* __restrict__ alpha, float* __restrict__ mbuf,
                             int E, int c, int Hc) {
  int idx = blockIdx.x * blockDim.x + threadIdx.x;
  if (idx >= E * NHEADS) return;
  int e = idx / NHEADS, h = idx % NHEADS;
  int s = src[e], d = dst[e];
  float ee = escale ? escale[(size_t)e * estride] : 0.f;
  const float* qp = qm + (size_t)d * Hc + h * c;
  const float* kp = km + (size_t)s * Hc + h * c;
  const float* wp = ew ? ew + h * c : nullptr;
  float a = 0.f;
  for (int j = 0; j < c; ++j) {
    float kv = kp[j] + (wp ? ee * wp[j] : 0.f);
    a += qp[j] * kv;
  }
  a *= rsqrtf((float)c);
  alpha[idx] = a;
  atomicMaxF(&mbuf[(size_t)d * NHEADS + h], a);
}

__global__ void k_edge_expsum(float* __restrict__ alpha, const int* __restrict__ dst,
                              const float* __restrict__ mbuf, float* __restrict__ sbuf, int E) {
  int idx = blockIdx.x * blockDim.x + threadIdx.x;
  if (idx >= E * NHEADS) return;
  int e = idx / NHEADS, h = idx % NHEADS;
  int d = dst[e];
  float m = mbuf[(size_t)d * NHEADS + h];
  if (!(m >= -1e37f)) m = 0.f;                 // isfinite() guard, as reference
  float a = __expf(alpha[idx] - m);
  alpha[idx] = a;
  atomicAdd(&sbuf[(size_t)d * NHEADS + h], a);
}

__global__ void k_edge_scatter(const float* __restrict__ vm,
                               const int* __restrict__ src, const int* __restrict__ dst,
                               const float* __restrict__ escale, int estride,
                               const float* __restrict__ ew,
                               const float* __restrict__ alpha, const float* __restrict__ sbuf,
                               float* __restrict__ out, int E, int c, int Hc) {
  int idx = blockIdx.x * blockDim.x + threadIdx.x;
  if (idx >= E * NHEADS) return;
  int e = idx / NHEADS, h = idx % NHEADS;
  int s = src[e], d = dst[e];
  float ee = escale ? escale[(size_t)e * estride] : 0.f;
  const float* wp = ew ? ew + h * c : nullptr;
  float w = alpha[idx] / (sbuf[(size_t)d * NHEADS + h] + 1e-16f);
  const float* vp = vm + (size_t)s * Hc + h * c;
  float* op = out + (size_t)d * Hc + h * c;
  for (int j = 0; j < c; ++j) {
    float vv = vp[j] + (wp ? ee * wp[j] : 0.f);
    atomicAdd(&op[j], vv * w);
  }
}

// beta = sigmoid([out|xr|out-xr] . betaw); y = beta*xr + (1-beta)*out
__global__ void k_beta_gate(const float* __restrict__ out, const float* __restrict__ xr,
                            const float* __restrict__ bw, float* __restrict__ y,
                            int Nn, int Hc) {
  int n = blockIdx.x * blockDim.x + threadIdx.x;
  if (n >= Nn) return;
  const float* o = out + (size_t)n * Hc;
  const float* r = xr + (size_t)n * Hc;
  float z = 0.f;
  for (int j = 0; j < Hc; ++j)
    z += bw[j] * o[j] + bw[Hc + j] * r[j] + bw[2 * Hc + j] * (o[j] - r[j]);
  float beta = 1.f / (1.f + __expf(-z));
  float* yp = y + (size_t)n * Hc;
  for (int j = 0; j < Hc; ++j) yp[j] = beta * r[j] + (1.f - beta) * o[j];
}

// ------------------------------ pooling -------------------------------------

__global__ void k_edge_gather_sum(const float* __restrict__ feat,
                                  const int* __restrict__ src, const int* __restrict__ dst,
                                  float* __restrict__ out, int E, int C) {
  size_t i = (size_t)blockIdx.x * blockDim.x + threadIdx.x;
  if (i >= (size_t)E * C) return;
  int e = (int)(i / C), ch = (int)(i % C);
  atomicAdd(&out[(size_t)dst[e] * C + ch], feat[(size_t)src[e] * C + ch]);
}

__global__ void k_pool_score(const float* __restrict__ agg, const float* __restrict__ nemb,
                             const float* __restrict__ wrel, const float* __restrict__ brel,
                             const float* __restrict__ wroot, float* __restrict__ score,
                             int Nn, int C) {
  int n = blockIdx.x * blockDim.x + threadIdx.x;
  if (n >= Nn) return;
  float s = brel[0];
  const float* a = agg + (size_t)n * C;
  const float* x = nemb + (size_t)n * C;
  for (int j = 0; j < C; ++j) s += a[j] * wrel[j] + x[j] * wroot[j];
  score[n] = s;
}

// per-frame top-k select (order-free since a max-pool follows) + max-pool
__global__ void k_topk_maxpool(const float* __restrict__ nemb, const float* __restrict__ score,
                               float* __restrict__ g, int C) {
  int f = blockIdx.x;
  __shared__ float sc[64];
  __shared__ float th[64];
  __shared__ int keep[64];
  int base = f * NPF;
  if (threadIdx.x < NPF) sc[threadIdx.x] = score[base + threadIdx.x];
  __syncthreads();
  if (threadIdx.x < NPF) {
    float s = sc[threadIdx.x];
    int rank = 0;
    for (int j = 0; j < NPF; ++j) {
      float sj = sc[j];
      if (sj > s || (sj == s && j < (int)threadIdx.x)) ++rank;
    }
    keep[threadIdx.x] = (rank < POOL_KP);
    th[threadIdx.x] = tanhf(s);
  }
  __syncthreads();
  for (int ch = threadIdx.x; ch < C; ch += blockDim.x) {
    float m = -3.0e38f;
    for (int j = 0; j < NPF; ++j)
      if (keep[j]) m = fmaxf(m, nemb[(size_t)(base + j) * C + ch] * th[j]);
    g[(size_t)f * C + ch] = m;
  }
}

// ------------------------- softmax / layernorm ------------------------------

__global__ void k_row_softmax(const float* __restrict__ in, float* __restrict__ out,
                              int rows, int cols, int ldin, int ldout) {
  int r = blockIdx.x;
  __shared__ float sh[256];
  float v = (threadIdx.x < (unsigned)cols) ? in[(size_t)r * ldin + threadIdx.x] : -3.0e38f;
  sh[threadIdx.x] = v;
  __syncthreads();
  for (int st = 128; st > 0; st >>= 1) {
    if (threadIdx.x < st) sh[threadIdx.x] = fmaxf(sh[threadIdx.x], sh[threadIdx.x + st]);
    __syncthreads();
  }
  float m = sh[0];
  __syncthreads();
  float e = (threadIdx.x < (unsigned)cols) ? __expf(v - m) : 0.f;
  sh[threadIdx.x] = e;
  __syncthreads();
  for (int st = 128; st > 0; st >>= 1) {
    if (threadIdx.x < st) sh[threadIdx.x] += sh[threadIdx.x + st];
    __syncthreads();
  }
  float s = sh[0];
  if (threadIdx.x < (unsigned)cols) out[(size_t)r * ldout + threadIdx.x] = e / s;
}

__global__ void k_layernorm(const float* __restrict__ X, const float* __restrict__ g,
                            const float* __restrict__ b, float* __restrict__ Y,
                            int ldy, int colOff, int d) {
  int r = blockIdx.x;
  __shared__ float ss[256], sq[256];
  float s = 0.f, q = 0.f;
  for (int j = threadIdx.x; j < d; j += blockDim.x) {
    float v = X[(size_t)r * d + j];
    s += v; q += v * v;
  }
  ss[threadIdx.x] = s; sq[threadIdx.x] = q;
  __syncthreads();
  for (int st = 128; st > 0; st >>= 1) {
    if (threadIdx.x < st) { ss[threadIdx.x] += ss[threadIdx.x + st]; sq[threadIdx.x] += sq[threadIdx.x + st]; }
    __syncthreads();
  }
  float m = ss[0] / (float)d;
  float var = sq[0] / (float)d - m * m;
  float rs = rsqrtf(var + 1e-5f);
  for (int j = threadIdx.x; j < d; j += blockDim.x) {
    float v = (X[(size_t)r * d + j] - m) * rs * g[j] + b[j];
    Y[(size_t)r * ldy + colOff + j] = v;
  }
}

// ======================= host-side orchestration ============================

static inline void gemm(hipStream_t st, const float* A, int lda,
                        const float* B, int ldb, int bNK,
                        const float* bias, const float* add, int ldadd,
                        float* C, int ldc, int colOff,
                        int M, int N, int K, float scale, int act) {
  dim3 grid((M + TBM - 1) / TBM, (N + TBN - 1) / TBN);
  k_gemm_bf16<<<grid, 256, 0, st>>>(A, lda, B, ldb, bNK, bias, add, ldadd,
                                    C, ldc, colOff, M, N, K, scale, act);
}

static inline void gemm_pre(hipStream_t st, const unsigned short* A, int lda,
                            const unsigned short* B, int ldb, const float* bias,
                            float* C, int ldc, int M, int N, int K, int act) {
  dim3 grid(M / TBM, N / TBN);
  k_gemm_bf16pre<<<grid, 256, 0, st>>>(A, lda, B, ldb, bias, C, ldc, M, N, K, act);
}

static inline void cvt_bf16(hipStream_t st, const float* X, int ldx,
                            unsigned short* Y, int ldy, int M, int K) {
  size_t n = (size_t)M * (K >> 3);
  k_cvt_bf16<<<(unsigned)((n + 255) / 256), 256, 0, st>>>(X, ldx, Y, ldy, M, K);
}

static inline void fillf(hipStream_t st, float* p, size_t n, float v) {
  k_fill<<<(unsigned)((n + 255) / 256), 256, 0, st>>>(p, n, v);
}

// Param leaf indexing: jax tree-flatten (alphabetical dict keys, depth-first).
enum {
  IN_X = 0, IN_EI, IN_IMG, IN_ATT, IN_EE, IN_TEI, IN_TW, IN_BATCH, IN_VEI,
  P_ATTEN_FC_B = 9, P_ATTEN_FC_W = 10,
  P_ENC_ATTEN = 11,   // 2 layers x 12 leaves
  P_ENC_GRAPH = 35,
  P_ENC_IMG = 59,
  P_FC1_B = 83, P_FC1_W = 84, P_FC2_B = 85, P_FC2_W = 86,
  P_GC1_SP = 87,      // beta.w, edge.w, k.b,k.w, q.b,q.w, skip.b,skip.w, v.b,v.w
  P_GC1_TM = 97,
  P_GC2_I3D = 107,    // beta.w, k.b,k.w, q.b,q.w, skip.b,skip.w, v.b,v.w
  P_GC2_SG = 116,
  P_IMG_FC_B = 125, P_IMG_FC_W = 126,
  P_OBJ_BN_B = 127, P_OBJ_BN_G = 128, P_OBJ_FC_B = 129, P_OBJ_FC_W = 130,
  P_POOL_REL_B = 131, P_POOL_REL_W = 132, P_POOL_ROOT_W = 133,
  P_X_BN_B = 134, P_X_BN_G = 135, P_X_FC_B = 136, P_X_FC_W = 137
};

static void run_tconv(hipStream_t st, const float* X, int Nn, int inD,
                      const int* src, const int* dst, int E, int c,
                      const float* escale, int estride,
                      void* const* din, int base, int hasEdge,
                      float* q, float* k, float* v, float* sk,
                      float* alpha, float* mbuf, float* sbuf,
                      float* attnout, float* gated) {
  const float* betaw = (const float*)din[base];
  const float* edgew = hasEdge ? (const float*)din[base + 1] : nullptr;
  int o = hasEdge ? 2 : 1;
  const float* kb = (const float*)din[base + o + 0];
  const float* kw = (const float*)din[base + o + 1];
  const float* qb = (const float*)din[base + o + 2];
  const float* qw = (const float*)din[base + o + 3];
  const float* sb = (const float*)din[base + o + 4];
  const float* sw = (const float*)din[base + o + 5];
  const float* vb = (const float*)din[base + o + 6];
  const float* vw = (const float*)din[base + o + 7];
  int Hc = NHEADS * c;
  gemm(st, X, inD, qw, inD, 1, qb, nullptr, 0, q,  Hc, 0, Nn, Hc, inD, 1.f, 0);
  gemm(st, X, inD, kw, inD, 1, kb, nullptr, 0, k,  Hc, 0, Nn, Hc, inD, 1.f, 0);
  gemm(st, X, inD, vw, inD, 1, vb, nullptr, 0, v,  Hc, 0, Nn, Hc, inD, 1.f, 0);
  gemm(st, X, inD, sw, inD, 1, sb, nullptr, 0, sk, Hc, 0, Nn, Hc, inD, 1.f, 0);
  fillf(st, mbuf, (size_t)Nn * NHEADS, -3.4e38f);
  fillf(st, sbuf, (size_t)Nn * NHEADS, 0.f);
  fillf(st, attnout, (size_t)Nn * Hc, 0.f);
  int eh = E * NHEADS;
  k_edge_alpha<<<(eh + 255) / 256, 256, 0, st>>>(q, k, src, dst, escale, estride,
                                                 edgew, alpha, mbuf, E, c, Hc);
  k_edge_expsum<<<(eh + 255) / 256, 256, 0, st>>>(alpha, dst, mbuf, sbuf, E);
  k_edge_scatter<<<(eh + 255) / 256, 256, 0, st>>>(v, src, dst, escale, estride,
                                                   edgew, alpha, sbuf, attnout, E, c, Hc);
  k_beta_gate<<<(Nn + 255) / 256, 256, 0, st>>>(attnout, sk, betaw, gated, Nn, Hc);
}

static void run_encoder(hipStream_t st, const float* xin, int T, int d, int nlayers,
                        void* const* din, int base,
                        float* qkv, float* scor, float* obuf, float* t1, float* t2,
                        float* ffb, float* xbuf,
                        float* out, int ldOut, int colOff) {
  const int dh = d / NHEADS;
  const float iscale = 1.f / sqrtf((float)dh);
  const float* cur = xin;
  for (int L = 0; L < nlayers; ++L) {
    int bi = base + L * 12; // ff1.b, ff1.w, ff2.b, ff2.w, in.b, in.w, ln1b, ln1g, ln2b, ln2g, out.b, out.w
    const float* ff1b = (const float*)din[bi + 0];
    const float* ff1w = (const float*)din[bi + 1];
    const float* ff2b = (const float*)din[bi + 2];
    const float* ff2w = (const float*)din[bi + 3];
    const float* inb  = (const float*)din[bi + 4];
    const float* inw  = (const float*)din[bi + 5];
    const float* ln1b = (const float*)din[bi + 6];
    const float* ln1g = (const float*)din[bi + 7];
    const float* ln2b = (const float*)din[bi + 8];
    const float* ln2g = (const float*)din[bi + 9];
    const float* outb = (const float*)din[bi + 10];
    const float* outw = (const float*)din[bi + 11];

    gemm(st, cur, d, inw, d, 1, inb, nullptr, 0, qkv, 3 * d, 0, T, 3 * d, d, 1.f, 0);
    for (int h = 0; h < NHEADS; ++h)   // scores_h = (q_h k_h^T)/sqrt(dh)
      gemm(st, qkv + h * dh, 3 * d, qkv + d + h * dh, 3 * d, 1, nullptr, nullptr, 0,
           scor + (size_t)h * T * T, T, 0, T, T, dh, iscale, 0);
    k_row_softmax<<<NHEADS * T, 256, 0, st>>>(scor, scor, NHEADS * T, T, T, T);
    for (int h = 0; h < NHEADS; ++h)   // o_h = att_h @ v_h  (B is [K,N])
      gemm(st, scor + (size_t)h * T * T, T, qkv + 2 * d + h * dh, 3 * d, 0,
           nullptr, nullptr, 0, obuf + h * dh, d, 0, T, dh, T, 1.f, 0);
    gemm(st, obuf, d, outw, d, 1, outb, cur, d, t1, d, 0, T, d, d, 1.f, 0);
    k_layernorm<<<T, 256, 0, st>>>(t1, ln1g, ln1b, t2, d, 0, d);
    gemm(st, t2, d, ff1w, d, 1, ff1b, nullptr, 0, ffb, 2048, 0, T, 2048, d, 1.f, 2);
    gemm(st, ffb, 2048, ff2w, 2048, 1, ff2b, t2, d, t1, d, 0, T, d, 2048, 1.f, 0);
    if (L == nlayers - 1) {
      k_layernorm<<<T, 256, 0, st>>>(t1, ln2g, ln2b, out, ldOut, colOff, d);
    } else {
      k_layernorm<<<T, 256, 0, st>>>(t1, ln2g, ln2b, xbuf, d, 0, d);
      cur = xbuf;
    }
  }
}

extern "C" void kernel_launch(void* const* d_in, const int* in_sizes, int n_in,
                              void* d_out, int out_size, void* d_ws, size_t ws_size,
                              hipStream_t stream) {
  (void)in_sizes; (void)n_in; (void)out_size; (void)ws_size;
  hipStream_t st = stream;
  const float* x    = (const float*)d_in[IN_X];
  const int*   ei   = (const int*)d_in[IN_EI];     // [0]=src row, [1]=dst row
  const float* img  = (const float*)d_in[IN_IMG];
  const float* attf = (const float*)d_in[IN_ATT];
  const float* ee   = (const float*)d_in[IN_EE];
  const int*   tei  = (const int*)d_in[IN_TEI];
  const float* tw   = (const float*)d_in[IN_TW];
  const int*   vei  = (const int*)d_in[IN_VEI];
#define PF(i) ((const float*)d_in[i])

  // --------- workspace layout (bump allocator, fp32 elements) ---------
  float* W = (float*)d_ws;
  size_t off = 0;
  auto alloc = [&](size_t n) { float* p = W + off; off += (n + 63) & ~(size_t)63; return p; };
  auto allocU = [&](size_t n) { return (unsigned short*)alloc((n + 1) / 2); };
  float* xfeat  = alloc((size_t)N_NODES * 256);
  float* xlabel = alloc((size_t)N_NODES * 64);
  float* xc     = alloc((size_t)N_NODES * 320);
  float* meanb  = alloc(2048);
  float* varb   = alloc(2048);
  float* qb_    = alloc((size_t)N_NODES * 256);
  float* kb_    = alloc((size_t)N_NODES * 256);
  float* vb_    = alloc((size_t)N_NODES * 256);
  float* skb_   = alloc((size_t)N_NODES * 256);
  float* alphab = alloc((size_t)E_SP * NHEADS);
  float* mbuf   = alloc((size_t)N_NODES * NHEADS);
  float* sbuf   = alloc((size_t)N_NODES * NHEADS);
  float* attn_o = alloc((size_t)N_NODES * 256);
  float* gated  = alloc((size_t)N_NODES * 256);
  float* nemb   = alloc((size_t)N_NODES * 512);
  float* aggb   = alloc((size_t)N_NODES * 512);
  float* score  = alloc((size_t)N_NODES);
  float* gemb   = alloc((size_t)N_FRAMES * 512);
  float* qkv    = alloc((size_t)N_FRAMES * 1536);
  float* scor   = alloc((size_t)NHEADS * N_FRAMES * N_FRAMES);
  float* obuf   = alloc((size_t)N_FRAMES * 512);
  float* t1     = alloc((size_t)N_FRAMES * 512);
  float* t2     = alloc((size_t)N_FRAMES * 512);
  float* ffb    = alloc((size_t)N_FRAMES * 2048);
  float* xbuf   = alloc((size_t)N_FRAMES * 512);
  float* imgp   = alloc((size_t)N_FRAMES * 256);
  float* imgt   = alloc((size_t)N_FRAMES * 256);
  float* attp   = alloc((size_t)N_FRAMES * 256);
  float* sgpre  = alloc((size_t)N_FRAMES * 512);
  float* fused  = alloc((size_t)N_FRAMES * 1792);
  float* fc1o   = alloc((size_t)N_FRAMES * 128);
  // bf16 staging for the K=2048 projections (async-DMA GEMM path)
  unsigned short* xbf   = allocU((size_t)N_NODES * INPUT_D);
  unsigned short* imgbf = allocU((size_t)N_FRAMES * 2048);
  unsigned short* attbf = allocU((size_t)N_FRAMES * 2048);
  unsigned short* wbf1  = allocU((size_t)256 * 2048);
  unsigned short* wbf2  = allocU((size_t)256 * 2048);
  unsigned short* wbf3  = allocU((size_t)256 * 2048);

  // 0) one-time fp32 -> bf16 conversions (halve re-read traffic, feed async GEMM)
  cvt_bf16(st, x, INPUT_D + OBJ_D, xbf, INPUT_D, N_NODES, INPUT_D);
  cvt_bf16(st, PF(P_X_FC_W), INPUT_D, wbf1, INPUT_D, 256, INPUT_D);
  cvt_bf16(st, img, 2048, imgbf, 2048, N_FRAMES, 2048);
  cvt_bf16(st, PF(P_IMG_FC_W), 2048, wbf2, 2048, 256, 2048);
  cvt_bf16(st, attf, 2048, attbf, 2048, N_FRAMES, 2048);
  cvt_bf16(st, PF(P_ATTEN_FC_W), 2048, wbf3, 2048, 256, 2048);

  // 1) node feature embeddings: lrelu(batchnorm(linear))
  gemm_pre(st, xbf, INPUT_D, wbf1, INPUT_D, PF(P_X_FC_B), xfeat, 256,
           N_NODES, 256, INPUT_D, 0);
  k_col_stats<<<256, 256, 0, st>>>(xfeat, 256, N_NODES, meanb, varb);
  k_norm_act<<<((size_t)N_NODES * 256 + 255) / 256, 256, 0, st>>>(
      xfeat, 256, meanb, varb, PF(P_X_BN_G), PF(P_X_BN_B), xc, 320, 0, N_NODES, 256, 1);
  gemm(st, x + INPUT_D, INPUT_D + OBJ_D, PF(P_OBJ_FC_W), OBJ_D, 1, PF(P_OBJ_FC_B),
       nullptr, 0, xlabel, 64, 0, N_NODES, 64, OBJ_D, 1.f, 0);
  k_col_stats<<<64, 256, 0, st>>>(xlabel, 64, N_NODES, meanb, varb);
  k_norm_act<<<((size_t)N_NODES * 64 + 255) / 256, 256, 0, st>>>(
      xlabel, 64, meanb, varb, PF(P_OBJ_BN_G), PF(P_OBJ_BN_B), xc, 320, 256, N_NODES, 64, 1);

  // 2) gc1_sp / gc1_tm TransformerConvs -> n_embed (concat, instnorm+lrelu)
  run_tconv(st, xc, N_NODES, 320, ei, ei + E_SP, E_SP, 64, ee + 3, 4,
            d_in, P_GC1_SP, 1, qb_, kb_, vb_, skb_, alphab, mbuf, sbuf, attn_o, gated);
  k_col_stats<<<256, 256, 0, st>>>(gated, 256, N_NODES, meanb, varb);
  k_norm_act<<<((size_t)N_NODES * 256 + 255) / 256, 256, 0, st>>>(
      gated, 256, meanb, varb, nullptr, nullptr, nemb, 512, 0, N_NODES, 256, 1);
  run_tconv(st, xc, N_NODES, 320, tei, tei + E_TM, E_TM, 64, tw, 1,
            d_in, P_GC1_TM, 1, qb_, kb_, vb_, skb_, alphab, mbuf, sbuf, attn_o, gated);
  k_col_stats<<<256, 256, 0, st>>>(gated, 256, N_NODES, meanb, varb);
  k_norm_act<<<((size_t)N_NODES * 256 + 255) / 256, 256, 0, st>>>(
      gated, 256, meanb, varb, nullptr, nullptr, nemb, 512, 256, N_NODES, 256, 1);

  // 3) SAGPooling score + per-frame top-k + global max pool -> g_embed
  fillf(st, aggb, (size_t)N_NODES * 512, 0.f);
  k_edge_gather_sum<<<((size_t)E_SP * 512 + 255) / 256, 256, 0, st>>>(
      nemb, ei, ei + E_SP, aggb, E_SP, 512);
  k_pool_score<<<(N_NODES + 255) / 256, 256, 0, st>>>(
      aggb, nemb, PF(P_POOL_REL_W), PF(P_POOL_REL_B), PF(P_POOL_ROOT_W), score, N_NODES, 512);
  k_topk_maxpool<<<N_FRAMES, 128, 0, st>>>(nemb, score, gemb, 512);

  // 4) graph encoder on g_embed -> fused[:, 0:512]
  run_encoder(st, gemb, N_FRAMES, 512, 2, d_in, P_ENC_GRAPH,
              qkv, scor, obuf, t1, t2, ffb, xbuf, fused, 1792, 0);

  // 5) image path: img_fc -> encoder -> img_trans -> gc2_i3d -> fused[:, 1024:1536]
  gemm_pre(st, imgbf, 2048, wbf2, 2048, PF(P_IMG_FC_B), imgp, 256,
           N_FRAMES, 256, 2048, 0);
  run_encoder(st, imgp, N_FRAMES, 256, 2, d_in, P_ENC_IMG,
              qkv, scor, obuf, t1, t2, ffb, xbuf, imgt, 256, 0);
  run_tconv(st, imgt, N_FRAMES, 256, vei, vei + E_VID, E_VID, 128, nullptr, 0,
            d_in, P_GC2_I3D, 0, qb_, kb_, vb_, skb_, alphab, mbuf, sbuf, attn_o, gated);
  k_col_stats<<<512, 256, 0, st>>>(gated, 512, N_FRAMES, meanb, varb);
  k_norm_act<<<((size_t)N_FRAMES * 512 + 255) / 256, 256, 0, st>>>(
      gated, 512, meanb, varb, nullptr, nullptr, fused, 1792, 1024, N_FRAMES, 512, 1);

  // 6) scene-graph path: gc2_sg on g_embed -> instnorm+lrelu -> encoder -> fused[:, 512:1024]
  run_tconv(st, gemb, N_FRAMES, 512, vei, vei + E_VID, E_VID, 128, nullptr, 0,
            d_in, P_GC2_SG, 0, qb_, kb_, vb_, skb_, alphab, mbuf, sbuf, attn_o, gated);
  k_col_stats<<<512, 256, 0, st>>>(gated, 512, N_FRAMES, meanb, varb);
  k_norm_act<<<((size_t)N_FRAMES * 512 + 255) / 256, 256, 0, st>>>(
      gated, 512, meanb, varb, nullptr, nullptr, sgpre, 512, 0, N_FRAMES, 512, 1);
  run_encoder(st, sgpre, N_FRAMES, 512, 2, d_in, P_ENC_GRAPH,
              qkv, scor, obuf, t1, t2, ffb, xbuf, fused, 1792, 512);

  // 7) attention path -> fused[:, 1536:1792]
  gemm_pre(st, attbf, 2048, wbf3, 2048, PF(P_ATTEN_FC_B), attp, 256,
           N_FRAMES, 256, 2048, 0);
  run_encoder(st, attp, N_FRAMES, 256, 2, d_in, P_ENC_ATTEN,
              qkv, scor, obuf, t1, t2, ffb, xbuf, fused, 1792, 1536);

  // 8) classifier head: fc1(lrelu) -> fc2 -> logits | softmax into d_out
  gemm(st, fused, 1792, PF(P_FC1_W), 1792, 1, PF(P_FC1_B), nullptr, 0,
       fc1o, 128, 0, N_FRAMES, 128, 1792, 1.f, 1);
  float* outp = (float*)d_out;
  gemm(st, fc1o, 128, PF(P_FC2_W), 128, 1, PF(P_FC2_B), nullptr, 0,
       outp, 2, 0, N_FRAMES, 2, 128, 1.f, 0);
  k_row_softmax<<<N_FRAMES, 256, 0, st>>>(outp, outp + (size_t)N_FRAMES * 2,
                                          N_FRAMES, 2, 2, 2);
#undef PF
}
